// PointNet2ClassPcModel_80659485819359
// MI455X (gfx1250) — matrix operand order repaired and checked by reading, not compiled, not run on hardware
//
#include <hip/hip_runtime.h>

typedef _Float16 half_t;
typedef __attribute__((ext_vector_type(16))) _Float16 v16h;
typedef __attribute__((ext_vector_type(8)))  _Float16 v8h;
typedef __attribute__((ext_vector_type(8)))  float    v8f;

#define EPSF 1e-5f

// ---------------------------------------------------------------------------
// Farthest point sampling: one block per batch, 256 threads, n/256 pts/thread.
// Records farthest BEFORE the min-update (matches the jax.lax.scan carry).
// ---------------------------------------------------------------------------
__global__ void fps_kernel(const float* __restrict__ xyz, int n, int npoint,
                           int* __restrict__ fidx, float* __restrict__ new_xyz) {
  int b = blockIdx.x, t = threadIdx.x;
  int nper = n >> 8; // n / 256  (n is 1024 or 512)
  const float* px = xyz + (size_t)b * n * 3;
  float dist[4];
  #pragma unroll
  for (int i = 0; i < 4; i++) dist[i] = 1e10f;
  __shared__ float s_val[256];
  __shared__ int   s_idx[256];
  __shared__ float s_c[3];
  __shared__ int   s_far;
  if (t == 0) s_far = 0;
  __syncthreads();
  for (int it = 0; it < npoint; it++) {
    if (t == 0) {
      int far = s_far;
      fidx[b * npoint + it] = far;
      float cx = px[far * 3 + 0], cy = px[far * 3 + 1], cz = px[far * 3 + 2];
      s_c[0] = cx; s_c[1] = cy; s_c[2] = cz;
      size_t o = ((size_t)b * npoint + it) * 3;
      new_xyz[o + 0] = cx; new_xyz[o + 1] = cy; new_xyz[o + 2] = cz;
    }
    __syncthreads();
    float cx = s_c[0], cy = s_c[1], cz = s_c[2];
    float best = -1.0f; int bi = 0;
    for (int i = 0; i < nper; i++) {
      int p = i * 256 + t;
      float dx = px[p * 3 + 0] - cx;
      float dy = px[p * 3 + 1] - cy;
      float dz = px[p * 3 + 2] - cz;
      float d = dx * dx + dy * dy + dz * dz;
      if (d < dist[i]) dist[i] = d;
      if (dist[i] > best) { best = dist[i]; bi = p; } // strict > => first max
    }
    s_val[t] = best; s_idx[t] = bi;
    __syncthreads();
    for (int o = 128; o > 0; o >>= 1) {
      if (t < o) {
        float v2 = s_val[t + o]; int i2 = s_idx[t + o];
        if (v2 > s_val[t] || (v2 == s_val[t] && i2 < s_idx[t])) {
          s_val[t] = v2; s_idx[t] = i2;
        }
      }
      __syncthreads();
    }
    if (t == 0) s_far = s_idx[0];
    __syncthreads();
  }
}

// ---------------------------------------------------------------------------
// Ball query: "mask + sort + take nsample" == first nsample in-radius indices
// in ascending order, padded with the first hit (center itself always hits).
// ---------------------------------------------------------------------------
__global__ void ball_query_kernel(const float* __restrict__ xyz,
                                  const float* __restrict__ new_xyz,
                                  int n, int S, int nsample, float r2,
                                  int total, int* __restrict__ gidx) {
  int c = blockIdx.x * blockDim.x + threadIdx.x;
  if (c >= total) return;
  int b = c / S;
  const float* px = xyz + (size_t)b * n * 3;
  float cx = new_xyz[c * 3 + 0], cy = new_xyz[c * 3 + 1], cz = new_xyz[c * 3 + 2];
  int* out = gidx + (size_t)c * nsample;
  int cnt = 0;
  for (int p = 0; p < n && cnt < nsample; p++) {
    float dx = px[p * 3 + 0] - cx;
    float dy = px[p * 3 + 1] - cy;
    float dz = px[p * 3 + 2] - cz;
    float d = dx * dx + dy * dy + dz * dz;
    if (d <= r2) out[cnt++] = p;
  }
  int f = (cnt > 0) ? out[0] : 0;
  for (; cnt < nsample; cnt++) out[cnt] = f;
}

// ---------------------------------------------------------------------------
// Build grouped rows (f16, channel-padded to CinP with zeros):
// [xyz[idx]-center (3) | feats[idx] (F) | 0-pad]
// ---------------------------------------------------------------------------
__global__ void gather_group_kernel(const float* __restrict__ xyz,
                                    const float* __restrict__ new_xyz,
                                    const half_t* __restrict__ feats,
                                    const int* __restrict__ gidx,
                                    int n, int S, int nsample, int F, int CinP,
                                    int total, half_t* __restrict__ out) {
  int g = blockIdx.x * blockDim.x + threadIdx.x;
  if (g >= total) return;
  int c = g / nsample;
  int b = c / S;
  int idx = gidx[g];
  const float* p = xyz + ((size_t)b * n + idx) * 3;
  half_t* row = out + (size_t)g * CinP;
  row[0] = (half_t)(p[0] - new_xyz[c * 3 + 0]);
  row[1] = (half_t)(p[1] - new_xyz[c * 3 + 1]);
  row[2] = (half_t)(p[2] - new_xyz[c * 3 + 2]);
  if (F > 0) {
    const half_t* f = feats + ((size_t)b * n + idx) * F;
    for (int k = 0; k < F; k++) row[3 + k] = f[k];
  }
  for (int k = 3 + F; k < CinP; k++) row[k] = (half_t)0.0f;
}

// SA3: concat raw xyz + feats per point, padded.
__global__ void concat_all_kernel(const float* __restrict__ xyz,
                                  const half_t* __restrict__ feats,
                                  int F, int CinP, int total,
                                  half_t* __restrict__ out) {
  int g = blockIdx.x * blockDim.x + threadIdx.x;
  if (g >= total) return;
  half_t* row = out + (size_t)g * CinP;
  row[0] = (half_t)xyz[g * 3 + 0];
  row[1] = (half_t)xyz[g * 3 + 1];
  row[2] = (half_t)xyz[g * 3 + 2];
  const half_t* f = feats + (size_t)g * F;
  for (int k = 0; k < F; k++) row[3 + k] = f[k];
  for (int k = 3 + F; k < CinP; k++) row[k] = (half_t)0.0f;
}

// Convert weights [Cout, Cin] f32 -> [Cout, CinP] f16 zero-padded.
__global__ void wconv_kernel(const float* __restrict__ W, int Cin, int CinP,
                             int total, half_t* __restrict__ Wh) {
  int i = blockIdx.x * blockDim.x + threadIdx.x;
  if (i >= total) return;
  int o = i / CinP, k = i % CinP;
  Wh[i] = (k < Cin) ? (half_t)W[(size_t)o * Cin + k] : (half_t)0.0f;
}

__global__ void zero_f32_kernel(float* __restrict__ p, int n) {
  int i = blockIdx.x * blockDim.x + threadIdx.x;
  if (i < n) p[i] = 0.0f;
}

// ---------------------------------------------------------------------------
// WMMA GEMM: Y[M,Cout](f16) = X[M,CinP](f16) @ Wh[Cout,CinP]^T + bias, plus
// f32 per-channel sum / sum-of-squares for the global batch-norm.
// One wave computes a 16x32 output tile (2 n-subtiles, reusing the A
// fragment => 2 WMMAs per A load, halves streaming-A traffic).
// 4 waves per block stacked over M.  v_wmma_f32_16x16x32_f16 lane mapping used here (k contiguous per half):
//   A lane L: row m0+(L%16); halves 0..15 = K k0+(L/16)*16.. (contig. X row)
//   B lane L: col n +(L%16); halves 0..15 = K k0+(L/16)*16.. (contig. W row)
//   D lane L: col n +(L%16); acc[r] = row m0 + r + (L/16)*8
// ---------------------------------------------------------------------------
__global__ void __launch_bounds__(128)
gemm_wmma_kernel(const half_t* __restrict__ X, const half_t* __restrict__ Wh,
                 const float* __restrict__ bias, half_t* __restrict__ Y,
                 float* __restrict__ s_sum, float* __restrict__ s_sq,
                 int M, int CinP, int Cout) {
  const int lane = threadIdx.x & 31;
  const int wave = threadIdx.x >> 5;
  const int n0   = blockIdx.x * 32;
  const int m0   = (blockIdx.y * 4 + wave) * 16;
  const int hsel = lane >> 4;  // 0: low lanes, 1: high lanes
  const int l16  = lane & 15;
  if (m0 >= M) return;  // wave-uniform; EXEC stays all-ones for WMMA

  const half_t* aptr  = X  + (size_t)(m0 + l16) * CinP + hsel * 16;
  const half_t* bptr0 = Wh + (size_t)(n0 + l16) * CinP + hsel * 16;
  const half_t* bptr1 = bptr0 + (size_t)16 * CinP;
  v8f acc[2] = {{}, {}};
  for (int k0 = 0; k0 < CinP; k0 += 32) {
    v8h a0 = *(const v8h*)(aptr + k0);
    v8h a1 = *(const v8h*)(aptr + k0 + 8);
    v16h af;
    #pragma unroll
    for (int i = 0; i < 8; i++) { af[i] = a0[i]; af[i + 8] = a1[i]; }
    v16h b0 = *(const v16h*)(bptr0 + k0);
    v16h b1 = *(const v16h*)(bptr1 + k0);
    acc[0] = __builtin_amdgcn_wmma_f32_16x16x32_f16(
        false, af, false, b0, (short)0, acc[0], false, false);
    acc[1] = __builtin_amdgcn_wmma_f32_16x16x32_f16(
        false, af, false, b1, (short)0, acc[1], false, false);
    // speculative stream-ahead prefetch of the A operand (~2KB ahead);
    // lowers to global_prefetch_b8, silently dropped if OOB.
    __builtin_prefetch(aptr + k0 + 1024, 0, 0);
  }
  #pragma unroll
  for (int u = 0; u < 2; u++) {
    const int ncol = n0 + u * 16 + l16;
    float bv = bias ? bias[ncol] : 0.0f;
    float s = 0.0f, q = 0.0f;
    #pragma unroll
    for (int r = 0; r < 8; r++) {
      float v = acc[u][r] + bv;
      int row = m0 + r + hsel * 8;
      Y[(size_t)row * Cout + ncol] = (half_t)v;
      s += v; q += v * v;
    }
    // lanes L and L^16 hold the same channel: combine, 16 atomics per subtile
    s += __shfl_xor(s, 16, 32);
    q += __shfl_xor(q, 16, 32);
    if (hsel == 0) {
      atomicAdd(&s_sum[ncol], s);
      atomicAdd(&s_sq[ncol], q);
    }
  }
}

// Per-channel affine from the accumulated stats: a = g*rsqrt(var+eps), sh = beta - mean*a
__global__ void finalize_stats_kernel(const float* __restrict__ s_sum,
                                      const float* __restrict__ s_sq,
                                      const float* __restrict__ gain,
                                      const float* __restrict__ beta,
                                      float invM, int C,
                                      float* __restrict__ a, float* __restrict__ sh) {
  int c = blockIdx.x * blockDim.x + threadIdx.x;
  if (c >= C) return;
  float m = s_sum[c] * invM;
  float v = s_sq[c] * invM - m * m;
  float ai = gain[c] * rsqrtf(v + EPSF);
  a[c] = ai;
  sh[c] = beta[c] - m * ai;
}

__global__ void affine_relu_kernel(const half_t* __restrict__ Y,
                                   const float* __restrict__ a,
                                   const float* __restrict__ sh,
                                   int total, int C, half_t* __restrict__ out) {
  int i = blockIdx.x * blockDim.x + threadIdx.x;
  if (i >= total) return;
  int c = i % C;
  float v = (float)Y[i] * a[c] + sh[c];
  out[i] = (half_t)fmaxf(v, 0.0f);
}

// Last layer of each SA block: relu(affine(y)) then max over the nsample axis.
__global__ void affine_relu_max_kernel(const half_t* __restrict__ Y,
                                       const float* __restrict__ a,
                                       const float* __restrict__ sh,
                                       int groups, int ns, int C,
                                       half_t* __restrict__ out) {
  int i = blockIdx.x * blockDim.x + threadIdx.x;
  if (i >= groups * C) return;
  int g = i / C, c = i % C;
  float ai = a[c], si = sh[c];
  const half_t* p = Y + ((size_t)g * ns) * C + c;
  float m = 0.0f;  // relu outputs are >= 0
  for (int j = 0; j < ns; j++) {
    float v = fmaxf((float)p[(size_t)j * C] * ai + si, 0.0f);
    m = fmaxf(m, v);
  }
  out[i] = (half_t)m;
}

// Final 32x40 head layer (no norm): plain f32 dot products, writes d_out.
__global__ void head_final_kernel(const half_t* __restrict__ H,
                                  const float* __restrict__ W3,
                                  const float* __restrict__ b3,
                                  float* __restrict__ out) {
  int i = blockIdx.x * blockDim.x + threadIdx.x;
  if (i >= 32 * 40) return;
  int m = i / 40, o = i % 40;
  const half_t* h = H + (size_t)m * 256;
  const float* w = W3 + (size_t)o * 256;
  float acc = b3[o];
  for (int c = 0; c < 256; c++) acc += (float)h[c] * w[c];
  out[i] = acc;
}

// ---------------------------------------------------------------------------
extern "C" void kernel_launch(void* const* d_in, const int* in_sizes, int n_in,
                              void* d_out, int out_size, void* d_ws, size_t ws_size,
                              hipStream_t stream) {
  (void)in_sizes; (void)n_in; (void)out_size; (void)ws_size;
  const float* points = (const float*)d_in[0];

  // layer tables: W input-index, real Cin, padded Cin (mult of 32), Cout
  const int Widx[11] = {1, 5, 9, 13, 17, 21, 25, 29, 33, 37, 41};
  const int Cin [11] = {3, 64, 64, 131, 128, 128, 259, 256, 512, 1024, 512};
  const int CinP[11] = {32, 64, 64, 160, 128, 128, 288, 256, 512, 1024, 512};
  const int Cout[11] = {64, 64, 128, 128, 128, 256, 256, 512, 1024, 512, 256};

  // ---- workspace bump allocator ----
  char* ws = (char*)d_ws;
  size_t off = 0;
  auto alloc = [&](size_t bytes) -> void* {
    void* p = ws + off;
    off = (off + bytes + 255) & ~(size_t)255;
    return p;
  };
  half_t* P0 = (half_t*)alloc((size_t)524288 * 128 * 2); // y ping (max 134 MB)
  half_t* P1 = (half_t*)alloc((size_t)262144 * 160 * 2); // act/grouped pong (84 MB)
  half_t* wh[11];
  for (int i = 0; i < 11; i++) wh[i] = (half_t*)alloc((size_t)Cout[i] * CinP[i] * 2);
  int*   fidx1 = (int*)  alloc((size_t)32 * 512 * 4);
  float* nxyz1 = (float*)alloc((size_t)32 * 512 * 3 * 4);
  int*   fidx2 = (int*)  alloc((size_t)32 * 128 * 4);
  float* nxyz2 = (float*)alloc((size_t)32 * 128 * 3 * 4);
  int*   gidx1 = (int*)  alloc((size_t)32 * 512 * 32 * 4);
  int*   gidx2 = (int*)  alloc((size_t)32 * 128 * 64 * 4);
  half_t* f1   = (half_t*)alloc((size_t)32 * 512 * 128 * 2);
  half_t* f2   = (half_t*)alloc((size_t)32 * 128 * 256 * 2);
  half_t* gbuf = (half_t*)alloc((size_t)32 * 1024 * 2);
  half_t* h1   = (half_t*)alloc((size_t)32 * 512 * 2);
  half_t* h2   = (half_t*)alloc((size_t)32 * 256 * 2);
  float* stats = (float*)alloc(2048 * 4); // sum | sumsq
  float* aff   = (float*)alloc(2048 * 4); // a | shift

  // ---- convert all weights to padded f16 once per call ----
  for (int i = 0; i < 11; i++) {
    int total = Cout[i] * CinP[i];
    wconv_kernel<<<(total + 255) / 256, 256, 0, stream>>>(
        (const float*)d_in[Widx[i]], Cin[i], CinP[i], total, wh[i]);
  }

  // gemm + global-norm stats + affine for layer li
  auto run_gemm = [&](const half_t* X, int li, int M) {
    zero_f32_kernel<<<8, 256, 0, stream>>>(stats, 2048);
    dim3 grid(Cout[li] / 32, (M + 63) / 64);
    gemm_wmma_kernel<<<grid, 128, 0, stream>>>(
        X, wh[li], (const float*)d_in[Widx[li] + 1], P0,
        stats, stats + 1024, M, CinP[li], Cout[li]);
    finalize_stats_kernel<<<(Cout[li] + 255) / 256, 256, 0, stream>>>(
        stats, stats + 1024,
        (const float*)d_in[Widx[li] + 2], (const float*)d_in[Widx[li] + 3],
        1.0f / (float)M, Cout[li], aff, aff + 1024);
  };
  auto act_full = [&](int li, int M, half_t* dst) {
    int total = M * Cout[li];
    affine_relu_kernel<<<(total + 255) / 256, 256, 0, stream>>>(
        P0, aff, aff + 1024, total, Cout[li], dst);
  };
  auto act_max = [&](int li, int groups, int ns, half_t* dst) {
    int total = groups * Cout[li];
    affine_relu_max_kernel<<<(total + 255) / 256, 256, 0, stream>>>(
        P0, aff, aff + 1024, groups, ns, Cout[li], dst);
  };

  // ================= SA1: 1024 pts -> 512 centers, r=0.2, ns=32 =============
  fps_kernel<<<32, 256, 0, stream>>>(points, 1024, 512, fidx1, nxyz1);
  ball_query_kernel<<<(16384 + 255) / 256, 256, 0, stream>>>(
      points, nxyz1, 1024, 512, 32, 0.04f, 16384, gidx1);
  gather_group_kernel<<<(524288 + 255) / 256, 256, 0, stream>>>(
      points, nxyz1, (const half_t*)nullptr, gidx1,
      1024, 512, 32, 0, 32, 524288, P1);
  const int M1 = 524288;
  run_gemm(P1, 0, M1); act_full(0, M1, P1);
  run_gemm(P1, 1, M1); act_full(1, M1, P1);
  run_gemm(P1, 2, M1); act_max(2, 16384, 32, f1);

  // ================= SA2: 512 pts -> 128 centers, r=0.4, ns=64 ==============
  fps_kernel<<<32, 256, 0, stream>>>(nxyz1, 512, 128, fidx2, nxyz2);
  ball_query_kernel<<<(4096 + 255) / 256, 256, 0, stream>>>(
      nxyz1, nxyz2, 512, 128, 64, 0.16f, 4096, gidx2);
  gather_group_kernel<<<(262144 + 255) / 256, 256, 0, stream>>>(
      nxyz1, nxyz2, f1, gidx2, 512, 128, 64, 128, 160, 262144, P1);
  const int M2 = 262144;
  run_gemm(P1, 3, M2); act_full(3, M2, P1);
  run_gemm(P1, 4, M2); act_full(4, M2, P1);
  run_gemm(P1, 5, M2); act_max(5, 4096, 64, f2);

  // ================= SA3 (group-all over 128 pts) ===========================
  concat_all_kernel<<<(4096 + 255) / 256, 256, 0, stream>>>(
      nxyz2, f2, 256, 288, 4096, P1);
  const int M3 = 4096;
  run_gemm(P1, 6, M3); act_full(6, M3, P1);
  run_gemm(P1, 7, M3); act_full(7, M3, P1);
  run_gemm(P1, 8, M3); act_max(8, 32, 128, gbuf);

  // ================= Head: 1024 -> 512 -> 256 -> 40 =========================
  run_gemm(gbuf, 9, 32);  act_full(9, 32, h1);
  run_gemm(h1, 10, 32);   act_full(10, 32, h2);
  head_final_kernel<<<(1280 + 255) / 256, 256, 0, stream>>>(
      h2, (const float*)d_in[45], (const float*)d_in[46], (float*)d_out);
}
